// Point_Transformer_Layer_33311766348350
// MI455X (gfx1250) — matrix operand order repaired and checked by reading, not compiled, not run on hardware
//
#include <hip/hip_runtime.h>
#include <hip/hip_bf16.h>

typedef __attribute__((ext_vector_type(16))) __bf16 v16bf;
typedef __attribute__((ext_vector_type(8)))  float  v8f;
typedef unsigned int u32v4 __attribute__((ext_vector_type(4)));
typedef int          i32v8 __attribute__((ext_vector_type(8)));
typedef int          i32v4 __attribute__((ext_vector_type(4)));

#define C_CH 128
#define NPTS 4096
#define BATCH 4
#define DQ 32

__device__ inline v8f wmma_bf16(v16bf a, v16bf b, v8f c) {
  return __builtin_amdgcn_wmma_f32_16x16x32_bf16(false, a, false, b, (short)0, c,
                                                 false, false);
}

// K index within a 16x32 bf16 A-tile for element e (0..15) and lane-half h (0/1)
__device__ inline int a_kidx(int e, int h) {
  return h * 16 + e;
}

// ---- Tensor Data Mover: async 2-D bf16 tile load (global -> LDS).
// D# per CDNA5 ISA ch.8: group0 {count,lds,global,type}, group1 {data_size=2B,
// tensor dims, tile dims, dim0 stride}. 6-arg builtin form.
__device__ inline void tdm_load_2d(unsigned lds_off, const void* gptr,
                                   unsigned td0, unsigned td1,
                                   unsigned tile0, unsigned tile1,
                                   unsigned stride0) {
  unsigned long long ga = (unsigned long long)(size_t)gptr;
  u32v4 g0 = {};
  g0[0] = 1u;                                                   // count=1, user D#
  g0[1] = lds_off;                                              // lds_addr (bytes)
  g0[2] = (unsigned)ga;                                         // global_addr[31:0]
  g0[3] = (unsigned)((ga >> 32) & 0x01FFFFFFull) | 0x80000000u; // addr[56:32]|type=2
  i32v8 g1 = {};
  g1[0] = (int)(1u << 16);                                      // data_size=1 -> 2B
  g1[1] = (int)((td0 & 0xFFFFu) << 16);                         // tensor_dim0 lo
  g1[2] = (int)(((td0 >> 16) & 0xFFFFu) | ((td1 & 0xFFFFu) << 16));
  g1[3] = (int)(((td1 >> 16) & 0xFFFFu) | ((tile0 & 0xFFFFu) << 16));
  g1[4] = (int)(tile1 & 0xFFFFu);                               // tile_dim1, dim2=0
  g1[5] = (int)stride0;                                         // dim0_stride[31:0]
  g1[6] = 0;
  g1[7] = 0;
  i32v4 g2 = {};
  i32v4 g3 = {};
  i32v8 g4 = {};
  __builtin_amdgcn_tensor_load_to_lds(g0, g1, g2, g3, g4, 0);
}

// ---------------- positional embedding
__global__ void k_pos(const float* __restrict__ xyz, const float* __restrict__ pw,
                      const float* __restrict__ pb, float* __restrict__ emb) {
  size_t i = (size_t)blockIdx.x * blockDim.x + threadIdx.x;
  if (i >= (size_t)BATCH * C_CH * NPTS) return;
  int n = (int)(i % NPTS);
  int c = (int)((i / NPTS) % C_CH);
  int b = (int)(i / ((size_t)C_CH * NPTS));
  const float* p = xyz + ((size_t)b * NPTS + n) * 3;
  emb[i] = pw[c * 3 + 0] * p[0] + pw[c * 3 + 1] * p[1] + pw[c * 3 + 2] * p[2] + pb[c];
}

// ---------------- OT[b,n,c] = bf16(A[b,c,n] + s*B[b,c,n])  (B may be null)
__global__ void k_makebfT(const float* __restrict__ A, const float* __restrict__ Bm,
                          float s, __bf16* __restrict__ OT) {
  size_t i = (size_t)blockIdx.x * blockDim.x + threadIdx.x;
  if (i >= (size_t)BATCH * C_CH * NPTS) return;
  int n = (int)(i % NPTS);
  int c = (int)((i / NPTS) % C_CH);
  int b = (int)(i / ((size_t)C_CH * NPTS));
  float v = A[i] + (Bm ? s * Bm[i] : 0.0f);
  OT[(size_t)b * NPTS * C_CH + (size_t)n * C_CH + c] = (__bf16)v;
}

// ---------------- WMMA GEMM, TDM-fed: Y[b,m,n] = sum_k W[m,k]*X[b,n,k] (+bias)
// XT layout [b][n][128] so the B gather is K-contiguous.
// grid (B, NPTS/64, M/16), block 128 (4 waves).
// dynamic LDS: Xs[2][64][32] bf16 (8 KB ping-pong, TDM destination)
__global__ void k_gemm(const float* __restrict__ W, const __bf16* __restrict__ XT,
                       const float* __restrict__ bias,
                       float* __restrict__ Y, __bf16* __restrict__ Ybf,
                       __bf16* __restrict__ YbfT, int M) {
  extern __shared__ char dynsm[];
  __bf16 (*Xs)[64][32] = (__bf16 (*)[64][32])dynsm;
  __shared__ __bf16 Ws[16][128];
  const int b = blockIdx.x, n0 = blockIdx.y * 64, m0 = blockIdx.z * 16;
  const int t = threadIdx.x, lane = t & 31, w = t >> 5;
  const int l15 = lane & 15, h = lane >> 4;
  const __bf16* Xb = XT + (size_t)b * NPTS * C_CH;
  for (int i = t; i < 16 * 128; i += 128) {
    int r = i >> 7, cc = i & 127;
    Ws[r][cc] = (__bf16)W[(size_t)(m0 + r) * C_CH + cc];
  }
  const unsigned gss = __builtin_amdgcn_groupstaticsize();
  if (w == 0)
    tdm_load_2d(gss, Xb + (size_t)n0 * C_CH, C_CH, NPTS, 32, 64, C_CH);
  v8f acc = {};
  for (int kc = 0; kc < 4; ++kc) {
    if (kc < 3) {
      if (w == 0) {
        tdm_load_2d(gss + (unsigned)(((kc + 1) & 1) * 4096),
                    Xb + (size_t)n0 * C_CH + (kc + 1) * 32, C_CH, NPTS, 32, 64, C_CH);
        __builtin_amdgcn_s_wait_tensorcnt(1);
      }
    } else if (w == 0) {
      __builtin_amdgcn_s_wait_tensorcnt(0);
    }
    __syncthreads();                 // tile kc resident (and Ws on first pass)
    v16bf a, bm;
#pragma unroll
    for (int e = 0; e < 16; ++e) {
      a[e]  = Ws[l15][kc * 32 + a_kidx(e, h)];
      bm[e] = Xs[kc & 1][w * 16 + l15][h * 16 + e];
    }
    acc = wmma_bf16(a, bm, acc);
    __syncthreads();                 // reads done before buffer reuse
  }
  const int col = n0 + w * 16 + l15;
#pragma unroll
  for (int r = 0; r < 8; ++r) {
    int m = m0 + r + 8 * h;
    float v = acc[r] + (bias ? bias[m] : 0.0f);
    if (Y)    Y[(size_t)b * M * NPTS + (size_t)m * NPTS + col] = v;
    if (Ybf)  Ybf[(size_t)b * M * NPTS + (size_t)m * NPTS + col] = (__bf16)v;
    if (YbfT) YbfT[(size_t)b * NPTS * M + (size_t)col * M + m] = (__bf16)v;
  }
}

// ---------------- attention pass 1: row softmax stats via symmetry (S == S^T)
// QT layout: [b][n][32] (K-contiguous for both WMMA operands).
// grid (B, NPTS/64), block 256. dynamic LDS: Qs[2][32][32] bf16 (TDM ping-pong)
__global__ void k_attn_stats(const __bf16* __restrict__ QT,
                             float* __restrict__ rowmax, float* __restrict__ rowsum) {
  extern __shared__ char dynsm[];
  __bf16 (*Qs)[32][32] = (__bf16 (*)[32][32])dynsm;
  __shared__ float red[8][2][16];
  __shared__ float cmax[64];
  __shared__ float csum[64];
  const int b = blockIdx.x, n0 = blockIdx.y * 64;
  const int t = threadIdx.x, lane = t & 31, w = t >> 5;
  const int nsub = w & 3, msub = w >> 2;
  const int l15 = lane & 15, h = lane >> 4;
  const __bf16* Qb = QT + (size_t)b * NPTS * DQ;
  const unsigned gss = __builtin_amdgcn_groupstaticsize();

  const int colB = n0 + nsub * 16 + l15;
  v16bf bq;                                  // B-matrix (d x n): contiguous in qT
#pragma unroll
  for (int e = 0; e < 16; ++e) bq[e] = Qb[(size_t)colB * DQ + h * 16 + e];

  // ---- sweep A: column max (== row max by symmetry)
  float rm = -3.0e38f;
  if (w == 0) tdm_load_2d(gss, Qb, DQ, NPTS, 32, 32, DQ);
  for (int mc = 0; mc < NPTS; mc += 32) {
    const int buf = (mc >> 5) & 1;
    if (mc + 32 < NPTS) {
      if (w == 0) {
        tdm_load_2d(gss + (unsigned)((buf ^ 1) * 2048),
                    Qb + (size_t)(mc + 32) * DQ, DQ, NPTS, 32, 32, DQ);
        __builtin_amdgcn_s_wait_tensorcnt(1);
      }
    } else if (w == 0) {
      __builtin_amdgcn_s_wait_tensorcnt(0);
    }
    __syncthreads();
    v16bf a;
#pragma unroll
    for (int e = 0; e < 16; ++e) a[e] = Qs[buf][msub * 16 + l15][a_kidx(e, h)];
    v8f s = {};
    s = wmma_bf16(a, bq, s);
#pragma unroll
    for (int r = 0; r < 8; ++r) rm = fmaxf(rm, s[r]);
    __syncthreads();
  }
  red[w][h][l15] = rm;
  __syncthreads();
  if (t < 64) {
    int ns = t >> 4, c16 = t & 15;
    cmax[t] = fmaxf(fmaxf(red[ns][0][c16], red[ns][1][c16]),
                    fmaxf(red[ns + 4][0][c16], red[ns + 4][1][c16]));
  }
  __syncthreads();

  // ---- sweep B: column sum of exp(s - max)
  const float cm = cmax[nsub * 16 + l15];
  float rs = 0.0f;
  if (w == 0) tdm_load_2d(gss, Qb, DQ, NPTS, 32, 32, DQ);
  for (int mc = 0; mc < NPTS; mc += 32) {
    const int buf = (mc >> 5) & 1;
    if (mc + 32 < NPTS) {
      if (w == 0) {
        tdm_load_2d(gss + (unsigned)((buf ^ 1) * 2048),
                    Qb + (size_t)(mc + 32) * DQ, DQ, NPTS, 32, 32, DQ);
        __builtin_amdgcn_s_wait_tensorcnt(1);
      }
    } else if (w == 0) {
      __builtin_amdgcn_s_wait_tensorcnt(0);
    }
    __syncthreads();
    v16bf a;
#pragma unroll
    for (int e = 0; e < 16; ++e) a[e] = Qs[buf][msub * 16 + l15][a_kidx(e, h)];
    v8f s = {};
    s = wmma_bf16(a, bq, s);
#pragma unroll
    for (int r = 0; r < 8; ++r) rs += __expf(s[r] - cm);
    __syncthreads();
  }
  red[w][h][l15] = rs;
  __syncthreads();
  if (t < 64) {
    int ns = t >> 4, c16 = t & 15;
    csum[t] = red[ns][0][c16] + red[ns][1][c16] +
              red[ns + 4][0][c16] + red[ns + 4][1][c16];
  }
  __syncthreads();
  if (t < 64) {
    rowmax[(size_t)b * NPTS + n0 + t] = cmax[t];
    rowsum[(size_t)b * NPTS + n0 + t] = csum[t];
  }
}

// ---------------- attention pass 2: x_r = V·A / (eps + colsum), fully fused
// grid (B, NPTS/64), block 256.
// dynamic LDS: Qs[2][32][32] @0 (4 KB), Vs[2][128][32] @4096 (16 KB), TDM-fed.
__global__ void k_attn_apply(const __bf16* __restrict__ QT, const __bf16* __restrict__ V,
                             const float* __restrict__ rowmax,
                             const float* __restrict__ rowsum,
                             float* __restrict__ XR) {
  extern __shared__ char dynsm[];
  __bf16 (*Qs)[32][32]  = (__bf16 (*)[32][32])dynsm;
  __bf16 (*Vs)[128][32] = (__bf16 (*)[128][32])(dynsm + 4096);
  __shared__ __bf16 AbufT[64][40];           // [m][n], K(n)-contiguous for B gathers
  __shared__ float rmaxs[32], rsums[32];
  __shared__ float red[8][2][16];
  __shared__ float csum[64];
  const int b = blockIdx.x, m0 = blockIdx.y * 64;
  const int t = threadIdx.x, lane = t & 31, w = t >> 5;
  const int nsub = w >> 2, msub = w & 3;
  const int l15 = lane & 15, h = lane >> 4;
  const __bf16* Qb = QT + (size_t)b * NPTS * DQ;
  const __bf16* Vb = V + (size_t)b * C_CH * NPTS;
  const unsigned gss = __builtin_amdgcn_groupstaticsize();

  const int colS = m0 + msub * 16 + l15;
  v16bf bq;                                  // B-matrix (d x m): contiguous in qT
#pragma unroll
  for (int e = 0; e < 16; ++e) bq[e] = Qb[(size_t)colS * DQ + h * 16 + e];

  v8f zero = {};
  v8f accY[4];
#pragma unroll
  for (int j = 0; j < 4; ++j) accY[j] = zero;
  float colacc = 0.0f;

  if (w == 0) {
    tdm_load_2d(gss, Qb, DQ, NPTS, 32, 32, DQ);
    tdm_load_2d(gss + 4096, Vb, NPTS, C_CH, 32, 128, NPTS);
  }
  for (int nc = 0; nc < NPTS; nc += 32) {
    const int buf = (nc >> 5) & 1, nb = buf ^ 1;
    if (nc + 32 < NPTS) {
      if (w == 0) {
        tdm_load_2d(gss + (unsigned)(nb * 2048),
                    Qb + (size_t)(nc + 32) * DQ, DQ, NPTS, 32, 32, DQ);
        tdm_load_2d(gss + 4096u + (unsigned)(nb * 8192), Vb + nc + 32,
                    NPTS, C_CH, 32, 128, NPTS);
        __builtin_amdgcn_s_wait_tensorcnt(2);
      }
    } else if (w == 0) {
      __builtin_amdgcn_s_wait_tensorcnt(0);
    }
    if (t < 32) {
      rmaxs[t] = rowmax[(size_t)b * NPTS + nc + t];
      rsums[t] = rowsum[(size_t)b * NPTS + nc + t];
    }
    __syncthreads();                 // tiles + row stats resident
    // S subtile (n-rows x m-cols) from LDS Q tile
    v16bf a;
#pragma unroll
    for (int e = 0; e < 16; ++e) a[e] = Qs[buf][nsub * 16 + l15][a_kidx(e, h)];
    v8f s = {};
    s = wmma_bf16(a, bq, s);
#pragma unroll
    for (int r = 0; r < 8; ++r) {
      int nl = nsub * 16 + r + 8 * h;
      float av = __expf(s[r] - rmaxs[nl]) / rsums[nl];
      colacc += av;
      AbufT[msub * 16 + l15][nl] = (__bf16)av;   // 8 consecutive n per lane
    }
    __syncthreads();                 // AbufT ready
    // V (c-tile x n-chunk) @ A (n-chunk x 64)
    v16bf va;
#pragma unroll
    for (int e = 0; e < 16; ++e) va[e] = Vs[buf][w * 16 + l15][a_kidx(e, h)];
#pragma unroll
    for (int j = 0; j < 4; ++j) {
      v16bf bm;
#pragma unroll
      for (int e = 0; e < 16; ++e) bm[e] = AbufT[j * 16 + l15][h * 16 + e];
      accY[j] = wmma_bf16(va, bm, accY[j]);
    }
    __syncthreads();                 // all LDS reads done before next TDM issue
  }

  red[w][h][l15] = colacc;
  __syncthreads();
  if (t < 64) {
    int ms = t >> 4, c16 = t & 15;
    csum[t] = 1e-9f + red[ms][0][c16] + red[ms][1][c16] +
              red[ms + 4][0][c16] + red[ms + 4][1][c16];
  }
  __syncthreads();
#pragma unroll
  for (int j = 0; j < 4; ++j) {
    float inv = 1.0f / csum[j * 16 + l15];
#pragma unroll
    for (int r = 0; r < 8; ++r) {
      int c = w * 16 + r + 8 * h;
      XR[(size_t)b * C_CH * NPTS + (size_t)c * NPTS + m0 + j * 16 + l15] =
          accY[j][r] * inv;
    }
  }
}

// ---------------- BN stats over (B, N) per channel
__global__ void k_bnstats(const float* __restrict__ Y, float* __restrict__ mean,
                          float* __restrict__ invstd) {
  const int c = blockIdx.x;
  __shared__ float s1[256], s2[256];
  float a = 0.0f, q = 0.0f;
  for (int i = threadIdx.x; i < BATCH * NPTS; i += 256) {
    int bb = i / NPTS, n = i % NPTS;
    float v = Y[(size_t)bb * C_CH * NPTS + (size_t)c * NPTS + n];
    a += v; q += v * v;
  }
  s1[threadIdx.x] = a; s2[threadIdx.x] = q;
  __syncthreads();
  for (int st = 128; st > 0; st >>= 1) {
    if (threadIdx.x < st) {
      s1[threadIdx.x] += s1[threadIdx.x + st];
      s2[threadIdx.x] += s2[threadIdx.x + st];
    }
    __syncthreads();
  }
  if (threadIdx.x == 0) {
    float m   = s1[0] / (float)(BATCH * NPTS);
    float var = s2[0] / (float)(BATCH * NPTS) - m * m;
    mean[c] = m;
    invstd[c] = rsqrtf(var + 1e-5f);
  }
}

// ---------------- fused BN + ReLU + residual + layer output + next-layer g/gbfT
__global__ void k_finalize(const float* __restrict__ Y, const float* __restrict__ mean,
                           const float* __restrict__ invstd,
                           const float* __restrict__ gam, const float* __restrict__ bet,
                           const float* __restrict__ emb,
                           float* __restrict__ g, __bf16* __restrict__ gbfT,
                           float* __restrict__ outp, int layer, int use_resid) {
  size_t i = (size_t)blockIdx.x * blockDim.x + threadIdx.x;
  if (i >= (size_t)BATCH * C_CH * NPTS) return;
  int n = (int)(i % NPTS);
  int c = (int)((i / NPTS) % C_CH);
  int b = (int)(i / ((size_t)C_CH * NPTS));
  float tv = (Y[i] - mean[c]) * invstd[c] * gam[c] + bet[c];
  tv = fmaxf(tv, 0.0f);
  float hn = (use_resid ? g[i] : 0.0f) + tv;
  if (outp)
    outp[(size_t)b * 5 * C_CH * NPTS + (size_t)(layer * C_CH + c) * NPTS + n] = hn;
  float gn = hn + emb[i];
  g[i] = gn;
  gbfT[(size_t)b * NPTS * C_CH + (size_t)n * C_CH + c] = (__bf16)gn;
}

extern "C" void kernel_launch(void* const* d_in, const int* in_sizes, int n_in,
                              void* d_out, int out_size, void* d_ws, size_t ws_size,
                              hipStream_t stream) {
  (void)in_sizes; (void)n_in; (void)out_size; (void)ws_size;
  const float* x      = (const float*)d_in[0];
  const float* xyz    = (const float*)d_in[1];
  const float* conv1w = (const float*)d_in[2];
  const float* bn1g   = (const float*)d_in[3];
  const float* bn1b   = (const float*)d_in[4];
  const float* posw   = (const float*)d_in[5];
  const float* posb   = (const float*)d_in[6];
  const float* wqk    = (const float*)d_in[7];
  const float* wv     = (const float*)d_in[8];
  const float* bv     = (const float*)d_in[9];
  const float* wt     = (const float*)d_in[10];
  const float* bt     = (const float*)d_in[11];
  const float* bng    = (const float*)d_in[12];
  const float* bnb    = (const float*)d_in[13];
  float* out = (float*)d_out;

  char* ws = (char*)d_ws;
  size_t off = 0;
  auto alloc = [&](size_t bytes) -> void* {
    void* p = ws + off;
    off = (off + bytes + 255) & ~(size_t)255;
    return p;
  };
  const size_t BCN = (size_t)BATCH * C_CH * NPTS;
  float*  emb  = (float*)alloc(BCN * 4);
  float*  g    = (float*)alloc(BCN * 4);
  float*  y    = (float*)alloc(BCN * 4);
  float*  xr   = (float*)alloc(BCN * 4);
  __bf16* gbfT = (__bf16*)alloc(BCN * 2);
  __bf16* vbf  = (__bf16*)alloc(BCN * 2);
  __bf16* tbfT = (__bf16*)alloc(BCN * 2);
  __bf16* qbfT = (__bf16*)alloc((size_t)BATCH * NPTS * DQ * 2);
  float*  rmx  = (float*)alloc((size_t)BATCH * NPTS * 4);
  float*  rsm  = (float*)alloc((size_t)BATCH * NPTS * 4);
  float*  mu   = (float*)alloc(C_CH * 4);
  float*  isd  = (float*)alloc(C_CH * 4);

  dim3 bl256(256);
  const int nblk = (int)((BCN + 255) / 256);

  k_pos<<<nblk, bl256, 0, stream>>>(xyz, posw, posb, emb);
  k_makebfT<<<nblk, bl256, 0, stream>>>(x, nullptr, 0.0f, gbfT);
  k_gemm<<<dim3(BATCH, NPTS / 64, C_CH / 16), dim3(128), 8192, stream>>>(
      conv1w, gbfT, nullptr, y, nullptr, nullptr, C_CH);
  k_bnstats<<<C_CH, bl256, 0, stream>>>(y, mu, isd);
  k_finalize<<<nblk, bl256, 0, stream>>>(y, mu, isd, bn1g, bn1b, emb, g, gbfT,
                                         nullptr, 0, 0);

  for (int L = 0; L < 5; ++L) {
    const float* wqkL = wqk + (size_t)L * DQ * C_CH;
    const float* wvL  = wv  + (size_t)L * C_CH * C_CH;
    const float* bvL  = bv  + (size_t)L * C_CH;
    const float* wtL  = wt  + (size_t)L * C_CH * C_CH;
    const float* btL  = bt  + (size_t)L * C_CH;
    const float* bngL = bng + (size_t)L * C_CH;
    const float* bnbL = bnb + (size_t)L * C_CH;

    k_gemm<<<dim3(BATCH, NPTS / 64, DQ / 16), dim3(128), 8192, stream>>>(
        wqkL, gbfT, nullptr, nullptr, nullptr, qbfT, DQ);
    k_gemm<<<dim3(BATCH, NPTS / 64, C_CH / 16), dim3(128), 8192, stream>>>(
        wvL, gbfT, bvL, nullptr, vbf, nullptr, C_CH);
    k_attn_stats<<<dim3(BATCH, NPTS / 64), bl256, 4096, stream>>>(qbfT, rmx, rsm);
    k_attn_apply<<<dim3(BATCH, NPTS / 64), bl256, 20480, stream>>>(qbfT, vbf, rmx,
                                                                   rsm, xr);
    k_makebfT<<<nblk, bl256, 0, stream>>>(g, xr, -1.0f, tbfT);
    k_gemm<<<dim3(BATCH, NPTS / 64, C_CH / 16), dim3(128), 8192, stream>>>(
        wtL, tbfT, btL, y, nullptr, nullptr, C_CH);
    k_bnstats<<<C_CH, bl256, 0, stream>>>(y, mu, isd);
    k_finalize<<<nblk, bl256, 0, stream>>>(y, mu, isd, bngL, bnbL, emb, g, gbfT,
                                           out, L, 1);
  }
}
